// Transformer_87402584473622
// MI455X (gfx1250) — matrix operand order repaired and checked by reading, not compiled, not run on hardware
//
#include <hip/hip_runtime.h>
#include <math.h>

// Problem constants (from reference)
#define BD   32
#define LD   400
#define HD   256
#define NHD  8
#define NLD  6
#define FF1D 128
#define FF2D 64

// GEMM tiling: 256 threads = 8 waves (wave32); 4 waves along M x 2 along N.
// Block tile 128x64, K-stage 64.  Each wave computes a 32x32 output tile
// (2x2 WMMA subtiles) -> 8 v_wmma per barrier pair.
#define TM 128
#define TN 64
#define TK 64

// gfx1250 async global->LDS staging (ASYNCcnt-tracked DMA-style loads)
#define USE_ASYNC_LDS 1

typedef __attribute__((ext_vector_type(16))) __bf16        v16bf;
typedef __attribute__((ext_vector_type(8)))  float         v8f;
typedef __attribute__((ext_vector_type(2)))  unsigned int  u32x2;
typedef __attribute__((ext_vector_type(4)))  unsigned int  u32x4;
typedef __attribute__((ext_vector_type(8)))  unsigned int  u32x8;

__device__ __forceinline__ unsigned short f2bf(float f) {
  unsigned int u = __float_as_uint(f);
  u += 0x7FFFu + ((u >> 16) & 1u);   // round-to-nearest-even
  return (unsigned short)(u >> 16);
}

#if USE_ASYNC_LDS
// GLOBAL_LOAD_ASYNC_TO_LDS_B128: per lane, copies 16 bytes from global vaddr to LDS offset vdst.
// Low 32 bits of a flat LDS pointer are the LDS byte offset (ISA 10.2).
__device__ __forceinline__ void async_ld128(unsigned lds_off, const void* gaddr) {
  asm volatile("global_load_async_to_lds_b128 %0, %1, off"
               :: "v"(lds_off), "v"(gaddr) : "memory");
}
__device__ __forceinline__ void wait_async0() {
  asm volatile("s_wait_asynccnt 0x0" ::: "memory");
}
#endif

// ---------------------------------------------------------------------------
// fp32 -> bf16 convert (grid-stride)
// ---------------------------------------------------------------------------
__global__ void k_f32_to_bf16(const float* __restrict__ src,
                              unsigned short* __restrict__ dst, long long n) {
  long long i  = (long long)blockIdx.x * blockDim.x + threadIdx.x;
  long long st = (long long)gridDim.x * blockDim.x;
  for (; i < n; i += st) dst[i] = f2bf(src[i]);
}

// ---------------------------------------------------------------------------
// x = x + pe[l, :]
// ---------------------------------------------------------------------------
__global__ void k_add_pe(const float* __restrict__ x, const float* __restrict__ pe,
                         float* __restrict__ out, long long n) {
  long long i  = (long long)blockIdx.x * blockDim.x + threadIdx.x;
  long long st = (long long)gridDim.x * blockDim.x;
  const long long LH = (long long)LD * HD;
  for (; i < n; i += st) out[i] = x[i] + pe[i % LH];
}

// ---------------------------------------------------------------------------
// masked_fill(att, pad_mask, -inf); mask shape (B, L, 1)
// ---------------------------------------------------------------------------
__global__ void k_mask(float* __restrict__ att, const unsigned char* __restrict__ mask,
                       long long n) {
  long long i  = (long long)blockIdx.x * blockDim.x + threadIdx.x;
  long long st = (long long)gridDim.x * blockDim.x;
  for (; i < n; i += st) {
    if (mask[i / HD]) att[i] = -INFINITY;
  }
}

// ---------------------------------------------------------------------------
// Softmax over the BATCH axis (reference uses legacy softmax(dim=0)).
// Per-head layout: sc[b*L*L + l*L + m].  One thread per (l,m); loop b in regs.
// Emits bf16 probabilities for the following WMMA GEMM.
// ---------------------------------------------------------------------------
__global__ void k_softmax_batch(const float* __restrict__ sc,
                                unsigned short* __restrict__ p, int lm) {
  int i  = blockIdx.x * blockDim.x + threadIdx.x;
  int st = gridDim.x * blockDim.x;
  for (; i < lm; i += st) {
    float v[BD];
    float mx = -INFINITY;
#pragma unroll
    for (int b = 0; b < BD; ++b) { v[b] = sc[(long long)b * lm + i]; mx = fmaxf(mx, v[b]); }
    float s = 0.f;
#pragma unroll
    for (int b = 0; b < BD; ++b) { v[b] = __expf(v[b] - mx); s += v[b]; }
    float inv = 1.f / s;
#pragma unroll
    for (int b = 0; b < BD; ++b) p[(long long)b * lm + i] = f2bf(v[b] * inv);
  }
}

// ---------------------------------------------------------------------------
// LayerNorm over H=256 with residual plumbing.
//   norm = LN(in)*g + b
//   out      = add_res_to_out ? norm + res_in : norm
//   out_bf   = bf16(out)                      (optional)
//   res_out  = res_in + norm                  (optional)
// One 256-thread block per row.
// ---------------------------------------------------------------------------
__global__ __launch_bounds__(HD) void k_layernorm(
    const float* __restrict__ in, const float* __restrict__ g,
    const float* __restrict__ bta, const float* __restrict__ res_in,
    float* __restrict__ out, unsigned short* __restrict__ out_bf,
    float* __restrict__ res_out, int add_res_to_out) {
  __shared__ float red[HD];
  const int row = blockIdx.x;
  const int t   = threadIdx.x;
  const long long base = (long long)row * HD;
  float x = in[base + t];

  red[t] = x;
  __syncthreads();
  for (int s = HD / 2; s > 0; s >>= 1) { if (t < s) red[t] += red[t + s]; __syncthreads(); }
  float mean = red[0] * (1.0f / HD);
  __syncthreads();
  float d = x - mean;
  red[t] = d * d;
  __syncthreads();
  for (int s = HD / 2; s > 0; s >>= 1) { if (t < s) red[t] += red[t + s]; __syncthreads(); }
  float var = red[0] * (1.0f / HD);

  float norm = d * rsqrtf(var + 1e-6f) * g[t] + bta[t];
  float r = res_in ? res_in[base + t] : 0.0f;
  float o = add_res_to_out ? (norm + r) : norm;
  out[base + t] = o;
  if (out_bf) out_bf[base + t] = f2bf(o);
  if (res_out) res_out[base + t] = r + norm;
}

// ---------------------------------------------------------------------------
// Batched bf16 GEMM via v_wmma_f32_16x16x32_bf16.
//   C[z] = act( scale * A[z] x op(B[z]) + bias )
// A: row-major MxK (bf16).  B: row-major KxN, or NxK if transB (bf16).
// Cf fp32 and/or Cb bf16 outputs (same ldC).
// Interior tiles stage LDS via GLOBAL_LOAD_ASYNC_TO_LDS_B128; edge tiles use
// guarded synchronous loads with zero-fill.
// ---------------------------------------------------------------------------
__global__ __launch_bounds__(256) void k_gemm_bf16(
    const unsigned short* __restrict__ A,
    const unsigned short* __restrict__ Bmat,
    const float* __restrict__ bias,
    float* __restrict__ Cf,
    unsigned short* __restrict__ Cb,
    int M, int N, int K, int ldA, int ldB, int ldC,
    int transB, float scale, int relu,
    long long sA, long long sB, long long sC) {
  __shared__ __align__(16) unsigned short As[TM][TK];   // 16 KB
  __shared__ __align__(16) unsigned short Bs[TN][TK];   // 8 KB, transposed: Bs[n][k]

  const int z = blockIdx.z;
  A    += (long long)z * sA;
  Bmat += (long long)z * sB;
  const long long cofs = (long long)z * sC;

  const int m0 = blockIdx.x * TM;
  const int n0 = blockIdx.y * TN;

  const bool mfull = (m0 + TM <= M);
  const bool nfull = (n0 + TN <= N);

  const int tid  = threadIdx.x;
  const int lane = tid & 31;
  const int wid  = tid >> 5;
  const int wm   = (wid & 3) * 32;   // wave M offset (2 subtiles of 16)
  const int wn   = (wid >> 2) * 32;  // wave N offset (2 subtiles of 16)

  const int half = lane >> 4;   // 0 | 1
  const int lr   = lane & 15;

  v8f acc[2][2] = {};

  const int ksteps = (K + TK - 1) / TK;
  for (int kt = 0; kt < ksteps; ++kt) {
    const int k0 = kt * TK;
    const bool kfull = (k0 + TK <= K);

    // ---- stage A tile: 128 rows x 64 k; each thread: 1 row, 32 contiguous bf16
    {
      const int r  = tid >> 1;          // 0..127
      const int kc = (tid & 1) * 32;    // 0 | 32
      const int gm = m0 + r;
      const unsigned short* gp = A + (long long)gm * ldA + k0 + kc;
#if USE_ASYNC_LDS
      if (mfull && kfull) {
#pragma unroll
        for (int c = 0; c < 4; ++c)
          async_ld128((unsigned)(size_t)&As[r][kc + c * 8], gp + c * 8);
      } else
#endif
      {
#pragma unroll
        for (int c = 0; c < 4; ++c) {
          const int gk = k0 + kc + c * 8;
          u32x4 val = {};
          if (gm < M && gk < K)         // all K's are multiples of 8
            val = *(const u32x4*)(gp + c * 8);
          *(u32x4*)&As[r][kc + c * 8] = val;
        }
        if (kt + 1 < ksteps && gm < M)
          __builtin_prefetch(gp + TK, 0, 1);
      }
    }
    // ---- stage B tile transposed into Bs[n][k]
    if (transB) {              // B stored [N][K]: straight coalesced copy
      const int n  = tid >> 2;          // 0..63
      const int kc = (tid & 3) * 16;    // 0..48
      const int gn = n0 + n;
      const unsigned short* gp = Bmat + (long long)gn * ldB + k0 + kc;
#if USE_ASYNC_LDS
      if (nfull && kfull) {
#pragma unroll
        for (int c = 0; c < 2; ++c)
          async_ld128((unsigned)(size_t)&Bs[n][kc + c * 8], gp + c * 8);
      } else
#endif
      {
#pragma unroll
        for (int c = 0; c < 2; ++c) {
          const int gk = k0 + kc + c * 8;
          u32x4 val = {};
          if (gn < N && gk < K)
            val = *(const u32x4*)(gp + c * 8);
          *(u32x4*)&Bs[n][kc + c * 8] = val;
        }
      }
    } else {                   // B stored [K][N]: transpose while staging
      const int k  = tid >> 2;          // 0..63
      const int nc = (tid & 3) * 16;    // 0..48
      const int gk = k0 + k;
#pragma unroll
      for (int c = 0; c < 2; ++c) {
        const int gn = n0 + nc + c * 8;
        u32x4 val = {};
        if (gk < K && gn < N)           // all N's are multiples of 8
          val = *(const u32x4*)(Bmat + (long long)gk * ldB + gn);
#pragma unroll
        for (int j = 0; j < 4; ++j) {
          Bs[nc + c * 8 + 2 * j + 0][k] = (unsigned short)(val[j] & 0xFFFFu);
          Bs[nc + c * 8 + 2 * j + 1][k] = (unsigned short)(val[j] >> 16);
        }
      }
    }
#if USE_ASYNC_LDS
    wait_async0();             // async LDS writes visible before the barrier
#endif
    __syncthreads();

    // ---- 2 K-substeps x (2 A-frags, 2 B-frags) -> 8 WMMAs per stage
#pragma unroll
    for (int ks = 0; ks < TK; ks += 32) {
      v16bf af[2], bf[2];
#pragma unroll
      for (int mi = 0; mi < 2; ++mi) {
        // A 16x32 bf16: lanes<16 -> row lr, K{0..7}U{16..23}; lanes>=16 -> K{8..15}U{24..31}
        const int row = wm + mi * 16 + lr;
        const int ca  = ks + half * 8;
        u32x4 lo = *(const u32x4*)&As[row][ca];
        u32x4 hi = *(const u32x4*)&As[row][ca + 16];
        u32x8 w;
        w[0] = lo[0]; w[1] = lo[1]; w[2] = lo[2]; w[3] = lo[3];
        w[4] = hi[0]; w[5] = hi[1]; w[6] = hi[2]; w[7] = hi[3];
        af[mi] = __builtin_bit_cast(v16bf, w);
      }
#pragma unroll
      for (int ni = 0; ni < 2; ++ni) {
        // B 32x16 bf16: lanes<16 -> col lr, K{0..7}U{16..23}; lanes>=16 -> K{8..15}U{24..31}
        const int col = wn + ni * 16 + lr;
        const int cb  = ks + half * 8;
        u32x4 lo = *(const u32x4*)&Bs[col][cb];
        u32x4 hi = *(const u32x4*)&Bs[col][cb + 16];
        u32x8 w;
        w[0] = lo[0]; w[1] = lo[1]; w[2] = lo[2]; w[3] = lo[3];
        w[4] = hi[0]; w[5] = hi[1]; w[6] = hi[2]; w[7] = hi[3];
        bf[ni] = __builtin_bit_cast(v16bf, w);
      }
#pragma unroll
      for (int mi = 0; mi < 2; ++mi)
#pragma unroll
        for (int ni = 0; ni < 2; ++ni)
          acc[mi][ni] = __builtin_amdgcn_wmma_f32_16x16x32_bf16(
              false, af[mi], false, bf[ni], (short)0, acc[mi][ni], false, false);
    }
    __syncthreads();
  }

  // ---- epilogue: D layout: VGPR r -> M = r + 8*half, N = lr
#pragma unroll
  for (int ni = 0; ni < 2; ++ni) {
    const int col = n0 + wn + ni * 16 + lr;
    if (col < N) {
      const float bval = bias ? bias[col] : 0.0f;
#pragma unroll
      for (int mi = 0; mi < 2; ++mi) {
#pragma unroll
        for (int r = 0; r < 8; ++r) {
          const int row = m0 + wm + mi * 16 + half * 8 + r;
          if (row < M) {
            float v = acc[mi][ni][r] * scale + bval;
            if (relu) v = fmaxf(v, 0.0f);
            const long long idx = cofs + (long long)row * ldC + col;
            if (Cf) Cf[idx] = v;
            if (Cb) Cb[idx] = f2bf(v);
          }
        }
      }
    }
  }
}

// ---------------------------------------------------------------------------
// Host orchestration
// ---------------------------------------------------------------------------
static inline size_t alignup256(size_t x) { return (x + 255) & ~(size_t)255; }

extern "C" void kernel_launch(void* const* d_in, const int* in_sizes, int n_in,
                              void* d_out, int out_size, void* d_ws, size_t ws_size,
                              hipStream_t stream) {
  (void)in_sizes; (void)n_in; (void)out_size; (void)ws_size;

  const float*         x_in = (const float*)d_in[0];
  const unsigned char* pad  = (const unsigned char*)d_in[1];
  const float*         pe   = (const float*)d_in[4];
  const float* Wq = (const float*)d_in[5];
  const float* bq = (const float*)d_in[6];
  const float* Wk = (const float*)d_in[7];
  const float* bk = (const float*)d_in[8];
  const float* Wv = (const float*)d_in[9];
  const float* bv = (const float*)d_in[10];
  const float* Wo = (const float*)d_in[11];
  const float* bo = (const float*)d_in[12];
  const float* W1 = (const float*)d_in[13];
  const float* b1 = (const float*)d_in[14];
  const float* W2 = (const float*)d_in[15];
  const float* b2 = (const float*)d_in[16];
  const float* W3 = (const float*)d_in[17];
  const float* b3 = (const float*)d_in[18];
  const float* ln_g = (const float*)d_in[19];
  const float* ln_b = (const float*)d_in[20];
  const float* fn_g = (const float*)d_in[21];
  const float* fn_b = (const float*)d_in[22];

  const long long nBLH = (long long)BD * LD * HD;   // 3,276,800
  const int M_ALL = BD * LD;                        // 12,800

  // ---- workspace carve-up
  char* ws = (char*)d_ws;
  size_t off = 0;
  auto take = [&](size_t bytes) { char* p = ws + off; off = alignup256(off + bytes); return p; };

  unsigned short* wq_bf = (unsigned short*)take((size_t)NLD * NHD * HD * HD * 2);
  unsigned short* wk_bf = (unsigned short*)take((size_t)NLD * NHD * HD * HD * 2);
  unsigned short* wv_bf = (unsigned short*)take((size_t)NLD * NHD * HD * HD * 2);
  unsigned short* wo_bf = (unsigned short*)take((size_t)NLD * NHD * HD * HD * 2);
  unsigned short* w1_bf = (unsigned short*)take((size_t)NLD * HD * FF1D * 2);
  unsigned short* w2_bf = (unsigned short*)take((size_t)NLD * FF1D * FF2D * 2);
  unsigned short* w3_bf = (unsigned short*)take((size_t)NLD * FF2D * HD * 2);

  float*          xbuf   = (float*)take((size_t)nBLH * 4);
  unsigned short* xbf    = (unsigned short*)take((size_t)nBLH * 2);
  unsigned short* qbf    = (unsigned short*)take((size_t)nBLH * 2);   // per-head [B*L, H]
  unsigned short* kbf    = (unsigned short*)take((size_t)nBLH * 2);
  unsigned short* vbf    = (unsigned short*)take((size_t)nBLH * 2);
  float*          scores = (float*)take((size_t)BD * LD * LD * 4);    // per-head [B, L, L]
  unsigned short* pbf    = (unsigned short*)take((size_t)BD * LD * LD * 2);
  unsigned short* attcat = (unsigned short*)take((size_t)M_ALL * NHD * HD * 2); // [B*L, 2048]
  float*          attout = (float*)take((size_t)nBLH * 4);
  float*          abuf   = (float*)take((size_t)nBLH * 4);
  unsigned short* abf    = (unsigned short*)take((size_t)nBLH * 2);
  float*          res2   = (float*)take((size_t)nBLH * 4);
  unsigned short* h1bf   = (unsigned short*)take((size_t)M_ALL * FF1D * 2);
  unsigned short* h2bf   = (unsigned short*)take((size_t)M_ALL * FF2D * 2);
  float*          fbuf   = (float*)take((size_t)nBLH * 4);

  auto conv = [&](const float* src, unsigned short* dst, long long n) {
    int blocks = (int)((n + 255) / 256); if (blocks > 4096) blocks = 4096;
    k_f32_to_bf16<<<blocks, 256, 0, stream>>>(src, dst, n);
  };
  auto gemm = [&](const unsigned short* A, const unsigned short* B, const float* bias,
                  float* Cf, unsigned short* Cb, int M, int N, int K,
                  int ldA, int ldB, int ldC, int transB, float scale, int relu,
                  int NZ, long long sA, long long sB, long long sC) {
    dim3 grid((M + TM - 1) / TM, (N + TN - 1) / TN, NZ);
    k_gemm_bf16<<<grid, 256, 0, stream>>>(A, B, bias, Cf, Cb, M, N, K,
                                          ldA, ldB, ldC, transB, scale, relu, sA, sB, sC);
  };

  // ---- weights -> bf16 (every call; deterministic, cheap)
  conv(Wq, wq_bf, (long long)NLD * NHD * HD * HD);
  conv(Wk, wk_bf, (long long)NLD * NHD * HD * HD);
  conv(Wv, wv_bf, (long long)NLD * NHD * HD * HD);
  conv(Wo, wo_bf, (long long)NLD * NHD * HD * HD);
  conv(W1, w1_bf, (long long)NLD * HD * FF1D);
  conv(W2, w2_bf, (long long)NLD * FF1D * FF2D);
  conv(W3, w3_bf, (long long)NLD * FF2D * HD);

  // ---- x = x + pe; xbf = bf16(x)
  k_add_pe<<<4096, 256, 0, stream>>>(x_in, pe, xbuf, nBLH);
  conv(xbuf, xbf, nBLH);

  const float att_scale = 1.0f / 16.0f;   // 1/sqrt(256)
  const int LL = LD * LD;

  for (int i = 0; i < NLD; ++i) {
    // ---------- attention ----------
    for (int h = 0; h < NHD; ++h) {
      const long long woff = ((long long)i * NHD + h) * HD * HD;
      const long long boff = ((long long)i * NHD + h) * HD;
      // q/k/v = x @ W + b  (bf16 out), M = B*L
      gemm(xbf, wq_bf + woff, bq + boff, nullptr, qbf,
           M_ALL, HD, HD, HD, HD, HD, 0, 1.0f, 0, 1, 0, 0, 0);
      gemm(xbf, wk_bf + woff, bk + boff, nullptr, kbf,
           M_ALL, HD, HD, HD, HD, HD, 0, 1.0f, 0, 1, 0, 0, 0);
      gemm(xbf, wv_bf + woff, bv + boff, nullptr, vbf,
           M_ALL, HD, HD, HD, HD, HD, 0, 1.0f, 0, 1, 0, 0, 0);
      // scores[b] = scale * q[b] @ k[b]^T   (fp32, batched over b)
      gemm(qbf, kbf, nullptr, scores, nullptr,
           LD, LD, HD, HD, HD, LD, 1, att_scale, 0,
           BD, (long long)LD * HD, (long long)LD * HD, (long long)LL);
      // softmax over batch axis -> bf16 probabilities
      k_softmax_batch<<<(LL + 255) / 256, 256, 0, stream>>>(scores, pbf, LL);
      // att[b] = p[b] @ v[b]; written directly to concat layout [B*L, h*H:(h+1)*H]
      gemm(pbf, vbf, nullptr, nullptr, attcat + (long long)h * HD,
           LD, HD, LD, LD, HD, NHD * HD, 0, 1.0f, 0,
           BD, (long long)LL, (long long)LD * HD, (long long)LD * NHD * HD);
    }
    // att @ Wo + bo  (fp32 out)
    gemm(attcat, wo_bf + (long long)i * NHD * HD * HD, bo + (long long)i * HD,
         attout, nullptr, M_ALL, HD, NHD * HD, NHD * HD, HD, HD, 0, 1.0f, 0, 1, 0, 0, 0);
    // masked_fill (pad_mask all-False, but honor semantics)
    k_mask<<<4096, 256, 0, stream>>>(attout, pad, nBLH);
    // a = LN(att); res2 = res1 + a; abf = bf16(a)
    k_layernorm<<<M_ALL, HD, 0, stream>>>(attout, ln_g + (long long)i * HD,
                                          ln_b + (long long)i * HD,
                                          xbuf /*res1*/, abuf, abf, res2, 0);
    // ---------- FFN (fed by a, per reference) ----------
    gemm(abf, w1_bf + (long long)i * HD * FF1D, b1 + (long long)i * FF1D,
         nullptr, h1bf, M_ALL, FF1D, HD, HD, FF1D, FF1D, 0, 1.0f, 1, 1, 0, 0, 0);
    gemm(h1bf, w2_bf + (long long)i * FF1D * FF2D, b2 + (long long)i * FF2D,
         nullptr, h2bf, M_ALL, FF2D, FF1D, FF1D, FF2D, FF2D, 0, 1.0f, 1, 1, 0, 0, 0);
    gemm(h2bf, w3_bf + (long long)i * FF2D * HD, b3 + (long long)i * HD,
         fbuf, nullptr, M_ALL, HD, FF2D, FF2D, HD, HD, 0, 1.0f, 0, 1, 0, 0, 0);
    // x = LN(f) + res2  (fp32 + bf16 mirror for next layer's projections)
    k_layernorm<<<M_ALL, HD, 0, stream>>>(fbuf, ln_g + (long long)i * HD,
                                          ln_b + (long long)i * HD,
                                          res2, xbuf, xbf, nullptr, 1);
  }

  // final LayerNorm -> d_out
  k_layernorm<<<M_ALL, HD, 0, stream>>>(xbuf, fn_g, fn_b, nullptr,
                                        (float*)d_out, nullptr, nullptr, 0);
}
